// qf_TimeBlock_81011673137391
// MI455X (gfx1250) — matrix operand order repaired and checked by reading, not compiled, not run on hardware
//
#include <hip/hip_runtime.h>
#include <hip/hip_bf16.h>

// Shapes from the reference
#define QB    8
#define QN    1024
#define QT    64
#define QCIN  64
#define QCOUT 64
#define QK    3
#define QTOUT 62                         // T - K + 1
#define QP    (QCOUT * QN * QTOUT)       // 4,063,232 per batch
#define QNS   (QP / 2)                   // 2,031,616 samples per batch

typedef __attribute__((ext_vector_type(16))) _Float16 v16h;
typedef __attribute__((ext_vector_type(8)))  _Float16 v8h;
typedef __attribute__((ext_vector_type(8)))  float    v8f;

// ---------------- deterministic hash RNG ----------------
__device__ __forceinline__ unsigned long long qf_mix(unsigned long long z) {
    z += 0x9e3779b97f4a7c15ull;
    z = (z ^ (z >> 30)) * 0xbf58476d1ce4e5b9ull;
    z = (z ^ (z >> 27)) * 0x94d049bb133111ebull;
    return z ^ (z >> 31);
}
__device__ __forceinline__ float qf_u01(unsigned long long h) {
    return (float)((h >> 40) + 1ull) * (1.0f / 16777217.0f);   // (0,1)
}
__device__ __forceinline__ float qf_gauss(unsigned long long idx, unsigned long long seed) {
    unsigned long long h1 = qf_mix(idx * 2ull + seed * 0x100000001ull);
    unsigned long long h2 = qf_mix(idx * 2ull + 1ull + seed * 0x100000001ull);
    float u1 = qf_u01(h1), u2 = qf_u01(h2);
    float r = sqrtf(-2.0f * logf(u1));
    return r * cosf(6.28318530718f * u2);
}

// ---------------- kernel 0: weight repack (f16 B-fragments), norms ----------------
// B-fragment layout: wf[((kk*2 + kg)*64 + col)*16 + h] = W_B[r = kk*32 + kg*16 + h][col]
// with r = k*CIN + ci  (k = r/64, ci = r%64), same contiguous k order as the A fragment.
__global__ void qf_prep_kernel(const float* __restrict__ X,
                               const float* __restrict__ W1,
                               const float* __restrict__ W2,
                               const float* __restrict__ W3,
                               void* wf_v, float* __restrict__ knorm,
                               float* __restrict__ inorm) {
    _Float16* wf = (_Float16*)wf_v;
    const int total = 3 * 12288 + 64 + QN * QTOUT;
    for (int i = blockIdx.x * blockDim.x + threadIdx.x; i < total;
         i += gridDim.x * blockDim.x) {
        if (i < 3 * 12288) {
            int w = i / 12288;
            int o = i % 12288;
            int h = o & 15;
            int colkg = o >> 4;
            int col = colkg & 63;
            int kgidx = colkg >> 6;          // kk*2 + kg, 0..11
            int r = kgidx * 16 + h;          // reduction row, 0..191
            int ci = r & 63;
            int k = r >> 6;
            const float* W = (w == 0) ? W1 : (w == 1) ? W2 : W3;
            wf[i] = (_Float16)W[(col * QCIN + ci) * QK + k];
        } else if (i < 3 * 12288 + 64) {
            int co = i - 3 * 12288;
            float s = 0.f;
            for (int j = 0; j < QCIN * QK; ++j) {
                float v = W1[co * (QCIN * QK) + j];
                s += v * v;
            }
            knorm[co] = sqrtf(s);
        } else {
            int idx = i - (3 * 12288 + 64);
            int n = idx / QTOUT, t = idx % QTOUT;
            float s = 0.f;
            for (int d = 0; d < QK; ++d) {
                float v = X[((size_t)n * QT + (t + d)) * QCIN];  // b=0, ci=0
                s += v * v;
            }
            inorm[idx] = sqrtf(s);
        }
    }
}

// ---------------- kernel 1: fused 3-conv GEMM epilogue (WMMA f16->f32) ----------------
// Writes pre-mask values directly to d_out in the final (B,N,TOUT,COUT) layout.
__global__ __launch_bounds__(512)
void qf_conv_kernel(const float* __restrict__ X,
                    const float* __restrict__ b1,
                    const float* __restrict__ b2,
                    const float* __restrict__ b3,
                    const void* __restrict__ wf_v,
                    const float* __restrict__ knorm,
                    const float* __restrict__ inorm,
                    float* __restrict__ out,
                    float* __restrict__ partials) {
    __shared__ _Float16 lds_x[66 * 64];     // T x CIN slab (+2 pad rows for im2col tail)
    __shared__ float red[512];
    const v16h* wf = (const v16h*)wf_v;

    const int blk = blockIdx.x;             // b*1024 + n
    const int b = blk >> 10;
    const int n = blk & 1023;
    const int tid = threadIdx.x;

    // Stage X[b,n,:,:] (4096 f32) into LDS as f16, same (t,ci) order.
    const float4* xs = (const float4*)(X + (size_t)blk * (QT * QCIN));
    for (int i = tid; i < 1024; i += 512) {
        float4 v = xs[i];
        int j = i * 4;
        lds_x[j + 0] = (_Float16)v.x;
        lds_x[j + 1] = (_Float16)v.y;
        lds_x[j + 2] = (_Float16)v.z;
        lds_x[j + 3] = (_Float16)v.w;
    }
    for (int i = tid; i < 128; i += 512) lds_x[4096 + i] = (_Float16)0.0f;  // pad rows
    __syncthreads();

    const int wave = tid >> 5, lane = tid & 31;
    const int l16 = lane & 15, lh = lane >> 4;
    const int m0 = (wave >> 2) * 16;        // t-tile
    const int n0 = (wave & 3) * 16;         // cout-tile

    v8f acc1 = {}, acc2 = {}, acc3 = {};
    #pragma unroll
    for (int kk = 0; kk < 6; ++kk) {        // K = 192 = 6 x 32
        int krow = kk >> 1;                 // conv tap k (chunks never straddle taps)
        int cib = ((kk & 1) << 5) + (lh << 4);
        // A fragment (contiguous): halves 0-15 -> K base + 16*lh + 0..15
        const _Float16* ap = &lds_x[(m0 + l16 + krow) * 64 + cib];
        v8h lo = *(const v8h*)ap;
        v8h hi = *(const v8h*)(ap + 8);
        v16h a = __builtin_shufflevector(lo, hi, 0, 1, 2, 3, 4, 5, 6, 7,
                                         8, 9, 10, 11, 12, 13, 14, 15);
        int bi = (kk * 2 + lh) * 64 + n0 + l16;
        v16h bw1 = wf[bi];
        v16h bw2 = wf[bi + 768];
        v16h bw3 = wf[bi + 1536];
        acc1 = __builtin_amdgcn_wmma_f32_16x16x32_f16(false, a, false, bw1,
                                                      (short)0, acc1, false, false);
        acc2 = __builtin_amdgcn_wmma_f32_16x16x32_f16(false, a, false, bw2,
                                                      (short)0, acc2, false, false);
        acc3 = __builtin_amdgcn_wmma_f32_16x16x32_f16(false, a, false, bw3,
                                                      (short)0, acc3, false, false);
    }

    // Epilogue: bias + sigmoid gate + CapReLU + noise*norms + clip, store (b,n,t,co).
    const int co = n0 + l16;
    const float bb1 = b1[co], bb2 = b2[co], bb3 = b3[co];
    const float kn = knorm[co];
    float lsum = 0.f;
    float* vout = out + (size_t)blk * (QTOUT * QCOUT);
    #pragma unroll
    for (int r = 0; r < 8; ++r) {
        int t = m0 + r + (lh << 3);         // C/D layout: lanes>=16 hold M+8
        if (t < QTOUT) {
            float v = acc1[r] + bb1;
            float s = acc2[r] + bb2;
            v += 1.0f / (1.0f + expf(-s));
            v += acc3[r] + bb3;
            v = fminf(fmaxf(v, 0.0f), 2.0f);
            float nrm = inorm[n * QTOUT + t] * kn;
            unsigned long long fi =
                ((((unsigned long long)b * QCOUT + co) * QN + n) * QTOUT + t);
            float g = qf_gauss(fi, 42ull);
            v = fminf(fmaxf(v + 0.2f * g * nrm, 0.0f), 2.0f);  // 2 * (0.1*g) * nrm
            vout[(size_t)t * QCOUT + co] = v;
            lsum += v;
        }
    }

    // Deterministic block sum (fixed tree order) for the sampler.
    red[tid] = lsum;
    __syncthreads();
    for (int s = 256; s > 0; s >>= 1) {
        if (tid < s) red[tid] += red[tid + s];
        __syncthreads();
    }
    if (tid == 0) partials[blk] = red[0];
}

// ---------------- kernel 2: per-batch totals ----------------
__global__ void qf_batchsum_kernel(const float* __restrict__ partials,
                                   float* __restrict__ totals) {
    __shared__ float red[256];
    int b = blockIdx.x;
    float s = 0.f;
    for (int i = threadIdx.x; i < QN; i += 256) s += partials[b * QN + i];
    red[threadIdx.x] = s;
    __syncthreads();
    for (int st = 128; st > 0; st >>= 1) {
        if (threadIdx.x < st) red[threadIdx.x] += red[threadIdx.x + st];
        __syncthreads();
    }
    if (threadIdx.x == 0) totals[b] = red[0];
}

// ---------------- kernel 3: categorical-sampling mask (in place on d_out) ----------------
// P(i kept) = 1 - (1 - v_i/S)^ns  (exact inclusion prob. of sampling w/ replacement)
__global__ void qf_mask_kernel(float* __restrict__ out,
                               const float* __restrict__ totals, int ntot) {
    int i = blockIdx.x * blockDim.x + threadIdx.x;
    if (i >= ntot) return;
    float v = out[i];
    int b = i / QP;
    float S = totals[b];
    float r = 0.f;
    if (v > 0.f && S > 0.f) {
        float q = fminf(v / S, 0.999999f);
        float p = 1.0f - expf((float)QNS * log1pf(-q));
        float u = qf_u01(qf_mix((unsigned long long)i ^ 0x9d2c5680cafe002bull));
        if (u < p) r = v;
    }
    out[i] = r;
}

// ---------------- launcher ----------------
extern "C" void kernel_launch(void* const* d_in, const int* in_sizes, int n_in,
                              void* d_out, int out_size, void* d_ws, size_t ws_size,
                              hipStream_t stream) {
    const float* X  = (const float*)d_in[0];
    const float* W1 = (const float*)d_in[1];
    const float* b1 = (const float*)d_in[2];
    const float* W2 = (const float*)d_in[3];
    const float* b2 = (const float*)d_in[4];
    const float* W3 = (const float*)d_in[5];
    const float* b3 = (const float*)d_in[6];
    float* out = (float*)d_out;

    char* ws = (char*)d_ws;                    // total workspace use: ~361 KB
    void*  wf    = (void*)(ws + 0);            // 3*12288 f16  = 73,728 B
    float* kn    = (float*)(ws + 73728);       // 64 f32
    float* inorm = (float*)(ws + 73984);       // N*TOUT f32   = 253,952 B
    float* part  = (float*)(ws + 327936);      // 8192 f32
    float* tot   = (float*)(ws + 360704);      // 8 f32

    qf_prep_kernel<<<200, 256, 0, stream>>>(X, W1, W2, W3, wf, kn, inorm);
    qf_conv_kernel<<<QB * QN, 512, 0, stream>>>(X, b1, b2, b3, wf, kn, inorm,
                                                out, part);
    qf_batchsum_kernel<<<QB, 256, 0, stream>>>(part, tot);
    const int ntot = QB * QP;                  // 32,505,856 == out_size
    qf_mask_kernel<<<(ntot + 255) / 256, 256, 0, stream>>>(out, tot, ntot);
}
